// Attention_19610820674341
// MI455X (gfx1250) — matrix operand order repaired and checked by reading, not compiled, not run on hardware
//
#include <hip/hip_runtime.h>
#include <math.h>

typedef __attribute__((ext_vector_type(16))) __bf16 v16bf;
typedef __attribute__((ext_vector_type(8)))  __bf16 v8bf;
typedef __attribute__((ext_vector_type(8)))  float  v8f;

#define B_  4
#define S_  2048
#define D_  1024
#define H_  16
#define HD_ 64
#define K_  1024

// ---------------------------------------------------------------------------
// fp32 -> bf16 conversion
// ---------------------------------------------------------------------------
__global__ void cvt_bf16(const float* __restrict__ in, __bf16* __restrict__ out, int n) {
    int i = blockIdx.x * blockDim.x + threadIdx.x;
    if (i < n) out[i] = (__bf16)in[i];
}

// ---------------------------------------------------------------------------
// LDS byte-address helper for inline asm (generic -> addrspace(3) offset).
// ---------------------------------------------------------------------------
__device__ __forceinline__ unsigned lds_addr_of(const void* p) {
    return (unsigned)(uintptr_t)(__attribute__((address_space(3))) const void*)p;
}

// ---------------------------------------------------------------------------
// A-fragment loader: 16x32 bf16, row-major source, 32-element row at `row0`.
// Lanes 0-15 hold K=0..15; lanes 16-31 hold K=16..31
// (row m = lane & 15).
// ---------------------------------------------------------------------------
__device__ __forceinline__ v16bf load_afrag(const __bf16* row0, int lane) {
    const int koff = (lane >> 4) * 16;
    v8bf lo = *(const v8bf*)(row0 + koff);
    v8bf hi = *(const v8bf*)(row0 + koff + 8);
    v16bf a;
#pragma unroll
    for (int e = 0; e < 8; ++e) { a[e] = lo[e]; a[8 + e] = hi[e]; }
    return a;
}

// ---------------------------------------------------------------------------
// GEMM: Y[m,n] = sum_k X[m,k] * W[n,k]   (X:[M,K] bf16, W:[N,K] bf16)
// One wave computes a 32x64 strip: 2 A-frags x 4 B-frags = 8 WMMAs per
// K-step. Ping-pong double buffering (two disjoint fragment sets, K-loop
// unrolled 2x) so there are no buffer-rotation register copies.
// mode 0: Q  -> rope, bf16 out [B,H,S,HD]
// mode 1: K  -> rope, bf16 out + fp32 out [B,H,S,HD]
// mode 2: V  ->       bf16 out + fp32 out [B,H,S,HD]
// mode 3: O  ->       fp32 out [B*S, D]
// ---------------------------------------------------------------------------
__global__ __launch_bounds__(32)
void gemm_xwT(const __bf16* __restrict__ X, const __bf16* __restrict__ W,
              float* __restrict__ outF, __bf16* __restrict__ outB,
              const float* __restrict__ cosT, const float* __restrict__ sinT,
              int mode) {
    const int lane    = threadIdx.x;
    const int half    = lane >> 4;
    const int ln      = lane & 15;
    const int tile_m  = blockIdx.y * 32;
    const int tile_n0 = blockIdx.x * 64;

    v8f c[2][4];
#pragma unroll
    for (int mi = 0; mi < 2; ++mi)
#pragma unroll
        for (int t = 0; t < 4; ++t)
#pragma unroll
            for (int v = 0; v < 8; ++v) c[mi][t][v] = 0.0f;

    const __bf16* arow0 = X + (size_t)(tile_m + ln) * K_;
    const __bf16* arow1 = X + (size_t)(tile_m + 16 + ln) * K_;
    const __bf16* brow[4];
#pragma unroll
    for (int t = 0; t < 4; ++t)
        brow[t] = W + (size_t)(tile_n0 + t * 16 + ln) * K_ + half * 16;

    // prologue: fill ping set (K-step 0)
    v16bf a0A = load_afrag(arow0, lane);
    v16bf a1A = load_afrag(arow1, lane);
    v16bf bA[4];
#pragma unroll
    for (int t = 0; t < 4; ++t) bA[t] = *(const v16bf*)(brow[t]);

    v16bf a0B, a1B, bB[4];

    for (int k0 = 0; k0 < K_; k0 += 64) {
        // load pong set (k0+32) while computing with ping set (k0)
        a0B = load_afrag(arow0 + k0 + 32, lane);
        a1B = load_afrag(arow1 + k0 + 32, lane);
#pragma unroll
        for (int t = 0; t < 4; ++t) bB[t] = *(const v16bf*)(brow[t] + k0 + 32);
        __builtin_prefetch(arow0 + k0 + 96, 0, 3);

#pragma unroll
        for (int t = 0; t < 4; ++t) {
            c[0][t] = __builtin_amdgcn_wmma_f32_16x16x32_bf16(
                          false, a0A, false, bA[t], (short)0, c[0][t], false, false);
            c[1][t] = __builtin_amdgcn_wmma_f32_16x16x32_bf16(
                          false, a1A, false, bA[t], (short)0, c[1][t], false, false);
        }

        // load ping set (k0+64, wraps harmlessly on last step) while
        // computing with pong set (k0+32)
        const int kn = (k0 + 64 < K_) ? (k0 + 64) : 0;
        a0A = load_afrag(arow0 + kn, lane);
        a1A = load_afrag(arow1 + kn, lane);
#pragma unroll
        for (int t = 0; t < 4; ++t) bA[t] = *(const v16bf*)(brow[t] + kn);

#pragma unroll
        for (int t = 0; t < 4; ++t) {
            c[0][t] = __builtin_amdgcn_wmma_f32_16x16x32_bf16(
                          false, a0B, false, bB[t], (short)0, c[0][t], false, false);
            c[1][t] = __builtin_amdgcn_wmma_f32_16x16x32_bf16(
                          false, a1B, false, bB[t], (short)0, c[1][t], false, false);
        }
    }

    // Epilogue. C layout: vgpr v, lanes0-15 -> row v, lanes16-31 -> row 8+v,
    // column = ln.
#pragma unroll
    for (int mi = 0; mi < 2; ++mi) {
#pragma unroll
        for (int t = 0; t < 4; ++t) {
            const int n = tile_n0 + t * 16 + ln;
#pragma unroll
            for (int v = 0; v < 8; ++v) {
                const int row  = tile_m + mi * 16 + v + half * 8;  // m in [0, B*S)
                const int spos = row & (S_ - 1);
                float val = c[mi][t][v];
                if (mode <= 1) {  // RoPE for Q and K
                    float p = __shfl_xor(val, 1);      // adjacent head-dim pair
                    int   i = (n & (HD_ - 1)) >> 1;
                    float cf = cosT[spos * (HD_ / 2) + i];
                    float sf = sinT[spos * (HD_ / 2) + i];
                    val = (n & 1) ? (p * sf + val * cf) : (val * cf - p * sf);
                }
                if (mode == 3) {
                    outF[(size_t)row * D_ + n] = val;
                } else {
                    const int bidx = row >> 11;        // row / S_
                    const int h    = n >> 6;           // n / HD_
                    const int dd   = n & (HD_ - 1);
                    const size_t idx =
                        (((size_t)(bidx * H_ + h)) * S_ + spos) * HD_ + dd;
                    outB[idx] = (__bf16)val;
                    if (outF) outF[idx] = val;
                }
            }
        }
    }
}

// ---------------------------------------------------------------------------
// Flash attention: one wave per (b, h, 16-query tile). Online softmax over
// 32-key tiles. QK^T and PV via v_wmma_f32_16x16x32_bf16. V tile staged in
// LDS with CDNA5 async global->LDS loads (ASYNCcnt).
// ---------------------------------------------------------------------------
__global__ __launch_bounds__(32)
void flash_attn(const __bf16* __restrict__ qb, const __bf16* __restrict__ kb,
                const __bf16* __restrict__ vb, __bf16* __restrict__ ab) {
    __shared__ __attribute__((aligned(16))) __bf16 pS[16 * 32];
    __shared__ __attribute__((aligned(16))) __bf16 vS[32 * 64];

    const int lane = threadIdx.x;
    const int half = lane >> 4;
    const int ln   = lane & 15;
    const int qt   = blockIdx.x;
    const int h    = blockIdx.y;
    const int b    = blockIdx.z;
    const size_t base = ((size_t)(b * H_ + h)) * S_ * HD_;
    const float scale = 0.125f;  // HD^-0.5

    const __bf16* qrow = qb + base + (size_t)(qt * 16 + ln) * HD_;
    v16bf aq0 = load_afrag(qrow, lane);        // d = 0..31
    v16bf aq1 = load_afrag(qrow + 32, lane);   // d = 32..63

    const unsigned vS_lane = lds_addr_of(&vS[0]) + (unsigned)lane * 128u;

    v8f o[4];
    float rowm[8], rowl[8];
#pragma unroll
    for (int t = 0; t < 4; ++t)
#pragma unroll
        for (int v = 0; v < 8; ++v) o[t][v] = 0.0f;
#pragma unroll
    for (int v = 0; v < 8; ++v) { rowm[v] = -INFINITY; rowl[v] = 0.0f; }

    const int ntiles = (qt * 16 + 16 + 31) / 32;
    for (int j = 0; j < ntiles; ++j) {
        const int kb0 = j * 32;

        // ---- async-stage V tile (32 keys x 64 d = 4KB) into LDS ----
        // Prior iteration's ds reads of vS must retire before the async
        // engine overwrites it.
        asm volatile("s_wait_dscnt 0x0" ::: "memory");
        {
            const uint64_t grow =
                (uint64_t)(uintptr_t)(vb + base + (size_t)(kb0 + lane) * HD_);
#pragma unroll
            for (int cidx = 0; cidx < 8; ++cidx) {
                unsigned la = vS_lane + (unsigned)cidx * 16u;
                uint64_t ga = grow + (uint64_t)cidx * 16u;
                asm volatile("global_load_async_to_lds_b128 %0, %1, off"
                             :: "v"(la), "v"(ga) : "memory");
            }
        }

        // ---- scores: two 16x16 tiles, K over HD=64 (two WMMAs each) ----
        v8f s[2];
#pragma unroll
        for (int t = 0; t < 2; ++t) {
#pragma unroll
            for (int v = 0; v < 8; ++v) s[t][v] = 0.0f;
            const __bf16* krow =
                kb + base + (size_t)(kb0 + t * 16 + ln) * HD_ + half * 16;
            v16bf bk0 = *(const v16bf*)krow;        // d = 0..31
            s[t] = __builtin_amdgcn_wmma_f32_16x16x32_bf16(
                       false, aq0, false, bk0, (short)0, s[t], false, false);
            v16bf bk1 = *(const v16bf*)(krow + 32); // d = 32..63
            s[t] = __builtin_amdgcn_wmma_f32_16x16x32_bf16(
                       false, aq1, false, bk1, (short)0, s[t], false, false);
        }

        // ---- online softmax (per-row over the 16-lane half) ----
#pragma unroll
        for (int v = 0; v < 8; ++v) {
            const int row = qt * 16 + v + half * 8;
            const int c0  = kb0 + ln;
            const int c1  = kb0 + 16 + ln;
            float x0 = (c0 <= row) ? s[0][v] * scale : -INFINITY;
            float x1 = (c1 <= row) ? s[1][v] * scale : -INFINITY;
            float mx = fmaxf(x0, x1);
#pragma unroll
            for (int off = 8; off >= 1; off >>= 1)
                mx = fmaxf(mx, __shfl_xor(mx, off));
            const float mnew  = fmaxf(rowm[v], mx);
            const float alpha = __expf(rowm[v] - mnew);
            const float p0 = __expf(x0 - mnew);
            const float p1 = __expf(x1 - mnew);
            float ls = p0 + p1;
#pragma unroll
            for (int off = 8; off >= 1; off >>= 1)
                ls += __shfl_xor(ls, off);
            rowl[v] = rowl[v] * alpha + ls;
            rowm[v] = mnew;
#pragma unroll
            for (int t = 0; t < 4; ++t) o[t][v] *= alpha;
            const int rl = v + half * 8;
            pS[rl * 32 + ln]      = (__bf16)p0;
            pS[rl * 32 + 16 + ln] = (__bf16)p1;
        }
        __syncthreads();  // single wave: orders pS store->load

        // ---- P (C-layout in LDS) -> A-fragment ----
        v16bf ap;
        {
            const int koff = half * 16;
#pragma unroll
            for (int e = 0; e < 8; ++e) {
                ap[e]     = pS[ln * 32 + koff + e];
                ap[8 + e] = pS[ln * 32 + koff + 8 + e];
            }
        }

        // async V tile must have landed before PV fragment reads
        asm volatile("s_wait_asynccnt 0x0" ::: "memory");

        // ---- PV: P(16x32) x V(32x64) ----
#pragma unroll
        for (int t = 0; t < 4; ++t) {
            v16bf bv;
#pragma unroll
            for (int e = 0; e < 16; ++e)
                bv[e] = vS[(half * 16 + e) * 64 + t * 16 + ln];
            o[t] = __builtin_amdgcn_wmma_f32_16x16x32_bf16(
                       false, ap, false, bv, (short)0, o[t], false, false);
        }
        __syncthreads();
    }

    // Normalize and write attention output, layout [B, S, D] with D = h*64+d.
#pragma unroll
    for (int t = 0; t < 4; ++t)
#pragma unroll
        for (int v = 0; v < 8; ++v) {
            const int q   = qt * 16 + v + half * 8;
            const float val = o[t][v] / rowl[v];
            ab[((size_t)(b * S_ + q)) * D_ + h * HD_ + t * 16 + ln] = (__bf16)val;
        }
}

// ---------------------------------------------------------------------------
// Host launch
// ---------------------------------------------------------------------------
extern "C" void kernel_launch(void* const* d_in, const int* in_sizes, int n_in,
                              void* d_out, int out_size, void* d_ws, size_t ws_size,
                              hipStream_t stream) {
    (void)in_sizes; (void)n_in; (void)out_size; (void)ws_size;

    const float* x    = (const float*)d_in[0];
    const float* cosT = (const float*)d_in[1];
    const float* sinT = (const float*)d_in[2];
    const float* wq   = (const float*)d_in[3];
    const float* wk   = (const float*)d_in[4];
    const float* wv   = (const float*)d_in[5];
    const float* wo   = (const float*)d_in[6];
    float* out = (float*)d_out;

    const size_t BSD = (size_t)B_ * S_ * D_;   // 8,388,608
    const size_t WSZ = (size_t)D_ * D_;        // 1,048,576

    char* ws = (char*)d_ws;
    __bf16* xb   = (__bf16*)(ws);                                  // 16 MB
    __bf16* wqb  = (__bf16*)(ws + (size_t)16 * 1024 * 1024);       //  2 MB
    __bf16* wkb  = wqb + WSZ;
    __bf16* wvb  = wkb + WSZ;
    __bf16* wob  = wvb + WSZ;
    __bf16* qbuf = (__bf16*)(ws + (size_t)24 * 1024 * 1024);       // 16 MB
    __bf16* kbuf = qbuf + BSD;
    __bf16* vbuf = kbuf + BSD;
    __bf16* abuf = vbuf + BSD;

    // 1) bf16 conversions
    cvt_bf16<<<(int)((BSD + 255) / 256), 256, 0, stream>>>(x, xb, (int)BSD);
    cvt_bf16<<<(int)((WSZ + 255) / 256), 256, 0, stream>>>(wq, wqb, (int)WSZ);
    cvt_bf16<<<(int)((WSZ + 255) / 256), 256, 0, stream>>>(wk, wkb, (int)WSZ);
    cvt_bf16<<<(int)((WSZ + 255) / 256), 256, 0, stream>>>(wv, wvb, (int)WSZ);
    cvt_bf16<<<(int)((WSZ + 255) / 256), 256, 0, stream>>>(wo, wob, (int)WSZ);

    // 2) Q/K/V projections (+RoPE on Q,K; K,V fp32 to d_out sections)
    dim3 ggrid(D_ / 64, (B_ * S_) / 32);  // (16, 256)
    gemm_xwT<<<ggrid, 32, 0, stream>>>(xb, wqb, nullptr,        qbuf, cosT, sinT, 0);
    gemm_xwT<<<ggrid, 32, 0, stream>>>(xb, wkb, out + BSD,      kbuf, cosT, sinT, 1);
    gemm_xwT<<<ggrid, 32, 0, stream>>>(xb, wvb, out + 2 * BSD,  vbuf, cosT, sinT, 2);

    // 3) causal flash attention
    flash_attn<<<dim3(S_ / 16, H_, B_), 32, 0, stream>>>(qbuf, kbuf, vbuf, abuf);

    // 4) output projection -> fp32 d_out[0 .. BSD)
    gemm_xwT<<<ggrid, 32, 0, stream>>>(abuf, wob, out, nullptr, nullptr, nullptr, 3);
}
